// InterpNet_49005576847465
// MI455X (gfx1250) — compile-verified, matrix operand order checked by reading, not run on hardware
//
#include <hip/hip_runtime.h>
#include <hip/hip_bf16.h>
#include <stdint.h>

// ---------------- problem constants ----------------
#define B_   4
#define H_   512
#define W_   512
#define PLANE ((size_t)H_ * W_)
#define NEG_ 0.1f
#define T_   0.5f

#define CIN1 38
#define XP   516      // H+4 (pad 2 halo) for conv1 input
#define HP   514      // H+2 (pad 1 halo) for conv2/conv3 inputs
#define K1R  950      // 38*25
#define K1   960      // padded to multiple of 64
#define K2   288      // 32*9
#define K2P  320      // padded to multiple of 64

typedef __bf16 bf16;
typedef __attribute__((ext_vector_type(16))) __bf16 v16bf;
typedef __attribute__((ext_vector_type(8)))  float  v8f;

// ---------------- workspace layout (bytes) ----------------
constexpr size_t OFF_FT1S = 0;                                   // B*2*H*W f32
constexpr size_t SZ_FT    = (size_t)B_ * 2 * PLANE * 4;
constexpr size_t OFF_FT2S = OFF_FT1S + SZ_FT;
constexpr size_t OFF_N1   = OFF_FT2S + SZ_FT;                    // B*H*W f32
constexpr size_t SZ_N     = (size_t)B_ * PLANE * 4;
constexpr size_t OFF_N2   = OFF_N1 + SZ_N;
constexpr size_t OFF_XPAD = OFF_N2 + SZ_N;                       // B*38*516*516 bf16
constexpr size_t SZ_XPAD  = (size_t)B_ * CIN1 * XP * XP * 2;
constexpr size_t OFF_H1   = OFF_XPAD + SZ_XPAD;                  // B*32*514*514 bf16
constexpr size_t SZ_H1    = (size_t)B_ * 32 * HP * HP * 2;
constexpr size_t OFF_H2   = OFF_H1 + SZ_H1;                      // B*16*514*514 bf16
constexpr size_t SZ_H2    = (size_t)B_ * 16 * HP * HP * 2;
constexpr size_t OFF_A1   = OFF_H2 + SZ_H2;                      // 32*960 bf16
constexpr size_t SZ_A1    = (size_t)32 * K1 * 2;
constexpr size_t OFF_A2   = OFF_A1 + SZ_A1;                      // 16*320 bf16
constexpr size_t SZ_A2    = (size_t)16 * K2P * 2;
constexpr size_t OFF_OT1  = OFF_A2 + SZ_A2;                      // 960 i32 im2col offsets
constexpr size_t SZ_OT1   = (size_t)K1 * 4;
constexpr size_t OFF_OT2  = OFF_OT1 + SZ_OT1;                    // 320 i32 im2col offsets
constexpr size_t SZ_OT2   = (size_t)K2P * 4;
constexpr size_t ZERO_END = OFF_OT2 + SZ_OT2;                    // zero everything above
constexpr size_t OFF_I1T  = ZERO_END;                            // B*3*H*W f32
constexpr size_t SZ_IT    = (size_t)B_ * 3 * PLANE * 4;
constexpr size_t OFF_I2T  = OFF_I1T + SZ_IT;

// ---------------- helpers ----------------
__device__ __forceinline__ v16bf load_frag(const bf16* rowp, int half) {
  // rowp = LDS row base ([row][64] bf16, 128B stride); lanes<16 take K {0..7,16..23},
  // lanes>=16 take K {8..15,24..31} relative to rowp -> two aligned 16B chunks.
  union { v16bf v; uint4 q[2]; } u;
  u.q[0] = *(const uint4*)(rowp + half);
  u.q[1] = *(const uint4*)(rowp + half + 16);
  return u.v;
}

// ---------------- kernel 0: zero workspace ----------------
__global__ void k_zero(uint32_t* __restrict__ p, size_t ndw) {
  size_t i = (size_t)blockIdx.x * blockDim.x + threadIdx.x;
  size_t stride = (size_t)gridDim.x * blockDim.x;
  for (; i < ndw; i += stride) p[i] = 0u;
}

// ---------------- kernel 1: forward splat (flow reversal) ----------------
__global__ void k_splat(const float* __restrict__ F1, const float* __restrict__ F2,
                        float* __restrict__ S1, float* __restrict__ N1,
                        float* __restrict__ S2, float* __restrict__ N2) {
  int idx = blockIdx.x * blockDim.x + threadIdx.x;
  if (idx >= B_ * H_ * W_) return;
  int x = idx % W_;
  int y = (idx / W_) % H_;
  int b = idx / (H_ * W_);
  size_t p = (size_t)y * W_ + x;
  for (int f = 0; f < 2; ++f) {
    const float* F = f ? F2 : F1;
    float* S = f ? S2 : S1;
    float* Nm = f ? N2 : N1;
    float u = F[((size_t)b * 2 + 0) * PLANE + p];
    float v = F[((size_t)b * 2 + 1) * PLANE + p];
    float X = (float)x + u, Y = (float)y + v;
    float x0f = floorf(X), y0f = floorf(Y);
    float fx = X - x0f, fy = Y - y0f;
    int x0 = (int)x0f, y0 = (int)y0f;
    const float wts[4] = {(1.f-fx)*(1.f-fy), fx*(1.f-fy), (1.f-fx)*fy, fx*fy};
    const int dxs[4] = {0,1,0,1}, dys[4] = {0,0,1,1};
    #pragma unroll
    for (int c4 = 0; c4 < 4; ++c4) {
      int xi = x0 + dxs[c4], yi = y0 + dys[c4];
      if (xi >= 0 && xi < W_ && yi >= 0 && yi < H_) {
        size_t lin = (size_t)yi * W_ + xi;
        float w = wts[c4];
        atomicAdd(&S[((size_t)b * 2 + 0) * PLANE + lin], u * w);
        atomicAdd(&S[((size_t)b * 2 + 1) * PLANE + lin], v * w);
        atomicAdd(&Nm[(size_t)b * PLANE + lin], w);
      }
    }
  }
}

// -------- kernel 2: weights f32 -> bf16 im2col matrices + offset tables --------
__global__ void k_wprep(const float* __restrict__ w1, const float* __restrict__ w2,
                        bf16* __restrict__ A1, bf16* __restrict__ A2,
                        int* __restrict__ ot1, int* __restrict__ ot2) {
  int i = blockIdx.x * blockDim.x + threadIdx.x;
  if (i < 32 * K1R) {
    int oc = i / K1R, k = i - oc * K1R;
    A1[(size_t)oc * K1 + k] = (bf16)w1[i];       // tail K1R..K1 stays zero
  }
  int j = i - 32 * K1R;
  if (j >= 0 && j < 16 * K2) {
    int oc = j / K2, k = j - oc * K2;
    A2[(size_t)oc * K2P + k] = (bf16)w2[j];      // tail K2..K2P stays zero
  }
  int t = j - 16 * K2;
  if (t >= 0 && t < K1) {
    int off = 0;
    if (t < K1R) {
      int c = t / 25, tap = t - c * 25, ky = tap / 5, kx = tap - ky * 5;
      off = (c * XP + ky) * XP + kx;
    }
    ot1[t] = off;
  }
  int t2 = t - K1;
  if (t2 >= 0 && t2 < K2P) {
    int off = 0;
    if (t2 < K2) {
      int c = t2 / 9, tap = t2 - c * 9, ky = tap / 3, kx = tap - ky * 3;
      off = (c * HP + ky) * HP + kx;
    }
    ot2[t2] = off;
  }
}

// ---------------- kernel 3: normalize flows, backwarp, pack bf16 x ----------------
__global__ void k_warp_pack(const float* __restrict__ I1, const float* __restrict__ I2,
                            const float* __restrict__ feat,
                            const float* __restrict__ S1, const float* __restrict__ N1,
                            const float* __restrict__ S2, const float* __restrict__ N2,
                            float* __restrict__ I1t, float* __restrict__ I2t,
                            bf16* __restrict__ xpad) {
  int idx = blockIdx.x * blockDim.x + threadIdx.x;
  if (idx >= B_ * H_ * W_) return;
  int x = idx % W_;
  int y = (idx / W_) % H_;
  int b = idx / (H_ * W_);
  size_t p = (size_t)y * W_ + x;
  for (int f = 0; f < 2; ++f) {
    const float* S = f ? S2 : S1;
    const float* Nm = f ? N2 : N1;
    const float* I = f ? I2 : I1;
    float* It = f ? I2t : I1t;
    float nv = Nm[(size_t)b * PLANE + p];
    float sx = S[((size_t)b * 2 + 0) * PLANE + p];
    float sy = S[((size_t)b * 2 + 1) * PLANE + p];
    float flx = (nv > 0.f) ? (-sx / nv) : -sx;
    float fly = (nv > 0.f) ? (-sy / nv) : -sy;
    // bilinear gather, align_corners=False, zero padding
    float xs = (float)x + flx - 0.5f;
    float ys = (float)y + fly - 0.5f;
    float x0f = floorf(xs), y0f = floorf(ys);
    float ax = xs - x0f, ay = ys - y0f;
    int x0 = (int)x0f, y0 = (int)y0f;
    float out0 = 0.f, out1 = 0.f, out2 = 0.f;
    const float wts[4] = {(1.f-ax)*(1.f-ay), ax*(1.f-ay), (1.f-ax)*ay, ax*ay};
    const int dxs[4] = {0,1,0,1}, dys[4] = {0,0,1,1};
    #pragma unroll
    for (int c4 = 0; c4 < 4; ++c4) {
      int xi = x0 + dxs[c4], yi = y0 + dys[c4];
      if (xi >= 0 && xi < W_ && yi >= 0 && yi < H_) {
        size_t lin = (size_t)yi * W_ + xi;
        float w = wts[c4];
        out0 += I[((size_t)b * 3 + 0) * PLANE + lin] * w;
        out1 += I[((size_t)b * 3 + 1) * PLANE + lin] * w;
        out2 += I[((size_t)b * 3 + 2) * PLANE + lin] * w;
      }
    }
    float outs[3] = {out0, out1, out2};
    #pragma unroll
    for (int c = 0; c < 3; ++c) {
      It[((size_t)b * 3 + c) * PLANE + p] = outs[c];
      xpad[(((size_t)b * CIN1 + f * 3 + c) * XP + (y + 2)) * XP + (x + 2)] = (bf16)outs[c];
    }
  }
  for (int c = 0; c < 32; ++c)
    xpad[(((size_t)b * CIN1 + 6 + c) * XP + (y + 2)) * XP + (x + 2)] =
        (bf16)feat[((size_t)b * 32 + c) * PLANE + p];
}

// ---------------- kernel 4: conv1 38->32 5x5 via WMMA bf16 ----------------
__global__ __launch_bounds__(256) void k_conv1(const bf16* __restrict__ xpad,
                                               const bf16* __restrict__ A1,
                                               const int* __restrict__ ot1,
                                               const float* __restrict__ b1,
                                               bf16* __restrict__ h1) {
  __shared__ bf16 As[32][64];   // [oc][k]
  __shared__ bf16 Bs[64][64];   // [pixel][k]
  int tid = threadIdx.x;
  int lane = tid & 31;
  int wave = tid >> 5;          // 0..7
  int waveM = wave & 1;         // oc half
  int waveN = wave >> 1;        // pixel 16-group
  int tile = blockIdx.x;        // B * H * (W/64)
  int x0 = (tile & 7) * 64;
  int y  = (tile >> 3) & (H_ - 1);
  int b  = tile >> 12;
  size_t xbase = (size_t)b * CIN1 * XP * XP + (size_t)y * XP + x0;

  v8f acc = {0.f, 0.f, 0.f, 0.f, 0.f, 0.f, 0.f, 0.f};
  int row  = lane & 15;
  int half = (lane < 16) ? 0 : 8;
  // staging roles (blockDim multiple of 64 -> per-thread k column is constant)
  int kB   = tid & 63;          // B k-column staged by this thread
  int pixB = tid >> 6;          // base pixel (0..3), step 4
  int dwA  = tid & 31;          // A dword within 64-elem row
  int rA   = tid >> 5;          // base A row (0..7), step 8

  for (int ks = 0; ks < K1 / 64; ++ks) {
    // --- stage A slice (32x64 bf16 = 1024 dwords) via async global->LDS DMA ---
    #pragma unroll
    for (int i = 0; i < 4; ++i) {
      int r = rA + i * 8;
      unsigned lds = (unsigned)(unsigned long long)&As[r][dwA * 2];
      unsigned go  = (unsigned)((r * K1 + ks * 64) * 2 + dwA * 4);
      asm volatile("global_load_async_to_lds_b32 %0, %1, %2"
                   :: "v"(lds), "v"(go), "s"(A1) : "memory");
    }
    // --- stage B slice: table-driven im2col gather (one table read per step) ---
    int kg  = ks * 64 + kB;
    int off = ot1[kg];                       // 0 for padded tail
    bool valid = kg < K1R;
    #pragma unroll
    for (int i = 0; i < 16; ++i) {
      int pix = pixB + i * 4;
      bf16 v = xpad[xbase + off + pix];
      if (!valid) v = (bf16)0.f;
      Bs[pix][kB] = v;
    }
    asm volatile("s_wait_asynccnt 0" ::: "memory");
    __syncthreads();

    v16bf a0 = load_frag(&As[16 * waveM + row][0], half);
    v16bf b0 = load_frag(&Bs[16 * waveN + row][0], half);
    acc = __builtin_amdgcn_wmma_f32_16x16x32_bf16(false, a0, false, b0,
                                                  (short)0, acc, false, false);
    v16bf a1 = load_frag(&As[16 * waveM + row][32], half);
    v16bf b1v = load_frag(&Bs[16 * waveN + row][32], half);
    acc = __builtin_amdgcn_wmma_f32_16x16x32_bf16(false, a1, false, b1v,
                                                  (short)0, acc, false, false);
    __syncthreads();
  }

  int px = x0 + 16 * waveN + (lane & 15);
  int ocHi = (lane < 16) ? 0 : 8;
  #pragma unroll
  for (int r = 0; r < 8; ++r) {
    int oc = 16 * waveM + r + ocHi;
    float v = acc[r] + b1[oc];
    v = (v > 0.f) ? v : NEG_ * v;
    h1[(((size_t)b * 32 + oc) * HP + (y + 1)) * HP + (px + 1)] = (bf16)v;
  }
}

// ---------------- kernel 5: conv2 32->16 3x3 via WMMA bf16 ----------------
__global__ __launch_bounds__(128) void k_conv2(const bf16* __restrict__ h1,
                                               const bf16* __restrict__ A2,
                                               const int* __restrict__ ot2,
                                               const float* __restrict__ b2,
                                               bf16* __restrict__ h2) {
  __shared__ bf16 As[16][64];
  __shared__ bf16 Bs[64][64];
  int tid = threadIdx.x;
  int lane = tid & 31;
  int waveN = tid >> 5;         // 0..3 pixel group
  int tile = blockIdx.x;
  int x0 = (tile & 7) * 64;
  int y  = (tile >> 3) & (H_ - 1);
  int b  = tile >> 12;
  size_t hbase = (size_t)b * 32 * HP * HP + (size_t)y * HP + x0;

  v8f acc = {0.f, 0.f, 0.f, 0.f, 0.f, 0.f, 0.f, 0.f};
  int row  = lane & 15;
  int half = (lane < 16) ? 0 : 8;
  int kB   = tid & 63;
  int pixB = tid >> 6;          // 0..1, step 2
  int dwA  = tid & 31;
  int rA   = tid >> 5;          // 0..3, step 4

  for (int ks = 0; ks < K2P / 64; ++ks) {
    #pragma unroll
    for (int i = 0; i < 4; ++i) {
      int r = rA + i * 4;
      unsigned lds = (unsigned)(unsigned long long)&As[r][dwA * 2];
      unsigned go  = (unsigned)((r * K2P + ks * 64) * 2 + dwA * 4);
      asm volatile("global_load_async_to_lds_b32 %0, %1, %2"
                   :: "v"(lds), "v"(go), "s"(A2) : "memory");
    }
    int kg  = ks * 64 + kB;
    int off = ot2[kg];
    bool valid = kg < K2;
    #pragma unroll
    for (int i = 0; i < 32; ++i) {
      int pix = pixB + i * 2;
      bf16 v = h1[hbase + off + pix];
      if (!valid) v = (bf16)0.f;
      Bs[pix][kB] = v;
    }
    asm volatile("s_wait_asynccnt 0" ::: "memory");
    __syncthreads();

    v16bf a0 = load_frag(&As[row][0], half);
    v16bf b0 = load_frag(&Bs[16 * waveN + row][0], half);
    acc = __builtin_amdgcn_wmma_f32_16x16x32_bf16(false, a0, false, b0,
                                                  (short)0, acc, false, false);
    v16bf a1 = load_frag(&As[row][32], half);
    v16bf b1v = load_frag(&Bs[16 * waveN + row][32], half);
    acc = __builtin_amdgcn_wmma_f32_16x16x32_bf16(false, a1, false, b1v,
                                                  (short)0, acc, false, false);
    __syncthreads();
  }

  int px = x0 + 16 * waveN + (lane & 15);
  int ocHi = (lane < 16) ? 0 : 8;
  #pragma unroll
  for (int r = 0; r < 8; ++r) {
    int oc = r + ocHi;
    float v = acc[r] + b2[oc];
    v = (v > 0.f) ? v : NEG_ * v;
    h2[(((size_t)b * 16 + oc) * HP + (y + 1)) * HP + (px + 1)] = (bf16)v;
  }
}

// ---------------- kernel 6: conv3 16->1 3x3 + sigmoid + blend ----------------
__global__ void k_conv3_blend(const bf16* __restrict__ h2,
                              const float* __restrict__ w3, const float* __restrict__ b3,
                              const float* __restrict__ I1t, const float* __restrict__ I2t,
                              float* __restrict__ out) {
  int idx = blockIdx.x * blockDim.x + threadIdx.x;
  if (idx >= B_ * H_ * W_) return;
  int x = idx % W_;
  int y = (idx / W_) % H_;
  int b = idx / (H_ * W_);
  size_t p = (size_t)y * W_ + x;
  float z = b3[0];
  #pragma unroll
  for (int c = 0; c < 16; ++c)
    #pragma unroll
    for (int ky = 0; ky < 3; ++ky)
      #pragma unroll
      for (int kx = 0; kx < 3; ++kx)
        z += (float)h2[(((size_t)b * 16 + c) * HP + (y + ky)) * HP + (x + kx)] *
             w3[(c * 3 + ky) * 3 + kx];
  float M = 1.f / (1.f + __expf(-z));
  float den = (1.f - T_) * M + T_ * (1.f - M);
  #pragma unroll
  for (int c = 0; c < 3; ++c) {
    float i1 = I1t[((size_t)b * 3 + c) * PLANE + p];
    float i2 = I2t[((size_t)b * 3 + c) * PLANE + p];
    float num = (1.f - T_) * M * i1 + T_ * (1.f - M) * i2;
    out[((size_t)b * 3 + c) * PLANE + p] = num / den;
  }
}

// ---------------- host launcher ----------------
extern "C" void kernel_launch(void* const* d_in, const int* in_sizes, int n_in,
                              void* d_out, int out_size, void* d_ws, size_t ws_size,
                              hipStream_t stream) {
  (void)in_sizes; (void)n_in; (void)out_size; (void)ws_size;
  const float* I1   = (const float*)d_in[0];
  const float* I2   = (const float*)d_in[1];
  const float* F1   = (const float*)d_in[2];
  const float* F2   = (const float*)d_in[3];
  const float* feat = (const float*)d_in[4];
  const float* w1   = (const float*)d_in[5];
  const float* b1   = (const float*)d_in[6];
  const float* w2   = (const float*)d_in[7];
  const float* b2   = (const float*)d_in[8];
  const float* w3   = (const float*)d_in[9];
  const float* b3   = (const float*)d_in[10];

  char* ws = (char*)d_ws;
  float* S1  = (float*)(ws + OFF_FT1S);
  float* S2  = (float*)(ws + OFF_FT2S);
  float* N1  = (float*)(ws + OFF_N1);
  float* N2  = (float*)(ws + OFF_N2);
  bf16*  xpad= (bf16*)(ws + OFF_XPAD);
  bf16*  h1  = (bf16*)(ws + OFF_H1);
  bf16*  h2  = (bf16*)(ws + OFF_H2);
  bf16*  A1  = (bf16*)(ws + OFF_A1);
  bf16*  A2  = (bf16*)(ws + OFF_A2);
  int*   ot1 = (int*)(ws + OFF_OT1);
  int*   ot2 = (int*)(ws + OFF_OT2);
  float* I1t = (float*)(ws + OFF_I1T);
  float* I2t = (float*)(ws + OFF_I2T);

  const int npix = B_ * H_ * W_;
  const int pixBlocks = (npix + 255) / 256;
  const int wprepWork = 32 * K1R + 16 * K2 + K1 + K2P;

  k_zero<<<4096, 256, 0, stream>>>((uint32_t*)ws, ZERO_END / 4);
  k_splat<<<pixBlocks, 256, 0, stream>>>(F1, F2, S1, N1, S2, N2);
  k_wprep<<<(wprepWork + 255) / 256, 256, 0, stream>>>(w1, w2, A1, A2, ot1, ot2);
  k_warp_pack<<<pixBlocks, 256, 0, stream>>>(I1, I2, feat, S1, N1, S2, N2, I1t, I2t, xpad);
  k_conv1<<<B_ * H_ * (W_ / 64), 256, 0, stream>>>(xpad, A1, ot1, b1, h1);
  k_conv2<<<B_ * H_ * (W_ / 64), 128, 0, stream>>>(h1, A2, ot2, b2, h2);
  k_conv3_blend<<<pixBlocks, 256, 0, stream>>>(h2, w3, b3, I1t, I2t, (float*)d_out);
}
